// Attention_69071664054717
// MI455X (gfx1250) — matrix operand order repaired and checked by reading, not compiled, not run on hardware
//
#include <hip/hip_runtime.h>
#include <stdint.h>

// Problem constants (from reference): B=2, T=2048, D=2048, N=8, K=1, H=256, G=8
#define B_   2
#define T_   2048
#define D_   2048
#define NH_  8
#define H_   256
#define NHD  2048        // N*H
#define QKV_COLS 2560    // N*H + 2*H  (q | k | v)

typedef __attribute__((ext_vector_type(16))) __bf16          v16bf;
typedef __attribute__((ext_vector_type(8)))  __bf16          v8bf;
typedef __attribute__((ext_vector_type(8)))  float           v8f;
typedef __attribute__((ext_vector_type(8)))  unsigned short  v8u;
typedef __attribute__((ext_vector_type(4)))  unsigned short  v4u;

union BF16x16 { v16bf v; v8bf h[2]; };

__device__ __forceinline__ unsigned short f2bf(float f) {
  unsigned int u = __float_as_uint(f);
  unsigned int r = u + 0x7fffu + ((u >> 16) & 1u);   // round-to-nearest-even
  return (unsigned short)(r >> 16);
}

// CDNA5 async DMA: global -> LDS, 16 bytes per lane, tracked by ASYNCcnt.
// VDST operand supplies the LDS byte address (low 32 bits of flat LDS ptr).
__device__ __forceinline__ void async_load_b128(unsigned int lds_off, const void* gaddr) {
  asm volatile("global_load_async_to_lds_b128 %0, %1, off"
               :: "v"(lds_off), "v"((unsigned long long)(uintptr_t)gaddr)
               : "memory");
}
__device__ __forceinline__ void wait_async0() {
  asm volatile("s_wait_asynccnt 0x0" ::: "memory");
}

// ---------------------------------------------------------------------------
// Kernel 1: f32 -> bf16 elementwise convert (vectorized x4)
// ---------------------------------------------------------------------------
__global__ __launch_bounds__(256) void cvt_f32_bf16(const float* __restrict__ in,
                                                    unsigned short* __restrict__ out,
                                                    int n4) {
  int i = blockIdx.x * blockDim.x + threadIdx.x;
  if (i >= n4) return;
  float4 f = ((const float4*)in)[i];
  v4u o;
  o[0] = f2bf(f.x); o[1] = f2bf(f.y); o[2] = f2bf(f.z); o[3] = f2bf(f.w);
  ((v4u*)out)[i] = o;
}

// ---------------------------------------------------------------------------
// Kernel 2: build concatenated QKV weight (D x 2560 bf16):
//   cols [0,2048)    = wq transposed: wcat[d][n*H+h] = wq[n][d][h]
//   cols [2048,2304) = wk:            wkv[0][0][d][h]
//   cols [2304,2560) = wv:            wkv[1][0][d][h]
// ---------------------------------------------------------------------------
__global__ __launch_bounds__(256) void build_wcat(const float* __restrict__ wq,
                                                  const float* __restrict__ wkv,
                                                  unsigned short* __restrict__ wcat) {
  int i = blockIdx.x * blockDim.x + threadIdx.x;
  if (i >= D_ * QKV_COLS) return;
  int d = i / QKV_COLS, c = i % QKV_COLS;
  float v;
  if (c < NHD) {
    int n = c >> 8, h = c & 255;
    v = wq[((size_t)n * D_ + d) * H_ + h];
  } else if (c < NHD + H_) {
    int h = c - NHD;
    v = wkv[(size_t)d * H_ + h];
  } else {
    int h = c - NHD - H_;
    v = wkv[(size_t)D_ * H_ + (size_t)d * H_ + h];
  }
  wcat[(size_t)d * QKV_COLS + c] = f2bf(v);
}

// ---------------------------------------------------------------------------
// Kernel 3: tiled bf16 GEMM, f32 out.  C[M,Nc] = A[M,Kd] * B[Kd,Nc]
// 256 threads = 8 waves; WG tile 64(M) x 128(N); wave tile 16x64 (4 WMMA/step
// with A-fragment reuse). K staged 32 deep, DOUBLE-BUFFERED in LDS via
// global_load_async_to_lds_b128 (ASYNCcnt), one barrier per K-step.
// ---------------------------------------------------------------------------
#define GA_STRIDE 40                 // A stage row stride (ushorts), 16B-aligned rows
#define GB_STRIDE 136                // B stage row stride (ushorts)
#define GA_STAGE  (64 * GA_STRIDE)   // 2560 ushorts
#define GB_STAGE  (32 * GB_STRIDE)   // 4352 ushorts

__global__ __launch_bounds__(256) void gemm_bf16_f32(const unsigned short* __restrict__ A,
                                                     const unsigned short* __restrict__ Bm,
                                                     float* __restrict__ C,
                                                     int M, int Nc, int Kd) {
  __shared__ unsigned short As[2 * GA_STAGE];   // 10,240 B
  __shared__ unsigned short Bs[2 * GB_STAGE];   // 17,408 B
  const int tid  = threadIdx.x;
  const int lane = tid & 31;
  const int wave = tid >> 5;
  const int r    = lane & 15;
  const int hl   = lane >> 4;
  const int m0   = (wave >> 1) * 16;            // wave M offset: 0,16,32,48
  const int n0   = (wave & 1) * 64;             // wave N offset: 0,64
  const int wgM  = blockIdx.y * 64;
  const int wgN  = blockIdx.x * 128;

  // Staging geometry: A tile 64x32 = 256 x 16B (1/thread); B 32x128 = 512 x 16B (2/thread)
  const int arow  = tid >> 2, achk = (tid & 3) * 8;
  const int bi0   = tid * 2, bi1 = tid * 2 + 1;
  const int brow0 = bi0 >> 4, bchk0 = (bi0 & 15) * 8;
  const int brow1 = bi1 >> 4, bchk1 = (bi1 & 15) * 8;

  auto issue_stage = [&](int buf, int kk) {
    async_load_b128((unsigned int)(uintptr_t)&As[buf * GA_STAGE + arow * GA_STRIDE + achk],
                    &A[(size_t)(wgM + arow) * Kd + kk + achk]);
    async_load_b128((unsigned int)(uintptr_t)&Bs[buf * GB_STAGE + brow0 * GB_STRIDE + bchk0],
                    &Bm[(size_t)(kk + brow0) * Nc + wgN + bchk0]);
    async_load_b128((unsigned int)(uintptr_t)&Bs[buf * GB_STAGE + brow1 * GB_STRIDE + bchk1],
                    &Bm[(size_t)(kk + brow1) * Nc + wgN + bchk1]);
  };

  v8f acc[4];
#pragma unroll
  for (int f = 0; f < 4; ++f)
#pragma unroll
    for (int i = 0; i < 8; ++i) acc[f][i] = 0.0f;

  issue_stage(0, 0);
  wait_async0();
  __syncthreads();

  for (int kk = 0; kk < Kd; kk += 32) {
    const int buf = (kk >> 5) & 1;
    if (kk + 32 < Kd) issue_stage(buf ^ 1, kk + 32);   // DMA next stage in background

    BF16x16 af;
    const int abase = buf * GA_STAGE + (m0 + r) * GA_STRIDE + 8 * hl;
    af.h[0] = *(const v8bf*)&As[abase];
    af.h[1] = *(const v8bf*)&As[abase + 16];
#pragma unroll
    for (int f = 0; f < 4; ++f) {
      BF16x16 bf;
      const int bbase = buf * GB_STAGE + n0 + f * 16 + r;
#pragma unroll
      for (int s = 0; s < 16; ++s) {
        const int k = (s < 8) ? (8 * hl + s) : (16 + 8 * hl + (s - 8));
        bf.v[s] = *(const __bf16*)&Bs[bbase + k * GB_STRIDE];
      }
      acc[f] = __builtin_amdgcn_wmma_f32_16x16x32_bf16(false, af.v, false, bf.v,
                                                       (short)0, acc[f], false, false);
    }
    wait_async0();
    __syncthreads();
  }

#pragma unroll
  for (int f = 0; f < 4; ++f)
#pragma unroll
    for (int i = 0; i < 8; ++i) {
      int row = wgM + m0 + i + 8 * hl;                 // C: lane=col, VGPR=row
      int col = wgN + n0 + f * 16 + r;
      C[(size_t)row * Nc + col] = acc[f][i];
    }
}

// ---------------------------------------------------------------------------
// Kernel 4: RoPE + split. One block per (b,t) row of the fused QKV f32 buffer.
// q: rope + H^-0.5 -> bf16; k: rope -> bf16; v: convert -> bf16.
// ---------------------------------------------------------------------------
__global__ __launch_bounds__(256) void rope_split(const float* __restrict__ qkvf,
                                                  const float* __restrict__ positions,
                                                  unsigned short* __restrict__ qb,
                                                  unsigned short* __restrict__ kb,
                                                  unsigned short* __restrict__ vb) {
  const int row = blockIdx.x;           // b*T + t
  const int tid = threadIdx.x;
  const float pos = positions[row];
  const size_t base = (size_t)row * QKV_COLS;

  // Q: 8 heads x 128 rotation pairs
  for (int idx = tid; idx < NH_ * 128; idx += 256) {
    int n = idx >> 7, h = idx & 127;
    float inv_ts = __powf(10000.0f, -(float)h * (1.0f / 128.0f));
    float rad = pos * inv_ts;
    float s, c;
    __sincosf(rad, &s, &c);
    float x1 = qkvf[base + n * H_ + h];
    float x2 = qkvf[base + n * H_ + 128 + h];
    qb[(size_t)row * NHD + n * H_ + h]       = f2bf((x1 * c - x2 * s) * 0.0625f);
    qb[(size_t)row * NHD + n * H_ + 128 + h] = f2bf((x2 * c + x1 * s) * 0.0625f);
  }
  // K: 128 pairs (single kv head)
  if (tid < 128) {
    int h = tid;
    float inv_ts = __powf(10000.0f, -(float)h * (1.0f / 128.0f));
    float rad = pos * inv_ts;
    float s, c;
    __sincosf(rad, &s, &c);
    float x1 = qkvf[base + NHD + h];
    float x2 = qkvf[base + NHD + 128 + h];
    kb[(size_t)row * H_ + h]       = f2bf(x1 * c - x2 * s);
    kb[(size_t)row * H_ + 128 + h] = f2bf(x2 * c + x1 * s);
  }
  // V
  vb[(size_t)row * H_ + tid] = f2bf(qkvf[base + NHD + H_ + tid]);
}

// ---------------------------------------------------------------------------
// Kernel 5: causal flash attention.
// Grid (T/64, N, B); 128 threads = 4 waves; wave owns 16 query rows x H=256.
// S-tile = 32 keys -> P tile is exactly one 16x32 bf16 A-fragment.
// ---------------------------------------------------------------------------
__global__ __launch_bounds__(128) void flash_attn(const unsigned short* __restrict__ qg,
                                                  const unsigned short* __restrict__ kg,
                                                  const unsigned short* __restrict__ vg,
                                                  unsigned short* __restrict__ encg) {
  __shared__ unsigned short KT[256 * 32];   // K transposed: [h][s], 16KB
  __shared__ unsigned short Vs[32 * 256];   // V row-major:  [s][h], 16KB
  __shared__ unsigned short Pw[4 * 16 * 32];// per-wave P staging, 4KB

  const int tid  = threadIdx.x;
  const int lane = tid & 31;
  const int wave = tid >> 5;
  const int r    = lane & 15;
  const int hl   = lane >> 4;
  const int t0   = blockIdx.x * 64;
  const int n    = blockIdx.y;
  const int bz   = blockIdx.z;
  const int wt0  = t0 + wave * 16;

  const unsigned short* qrow = qg + (size_t)(bz * T_ + wt0 + r) * NHD + n * H_;

  v8f O[16];
#pragma unroll
  for (int f = 0; f < 16; ++f)
#pragma unroll
    for (int i = 0; i < 8; ++i) O[f][i] = 0.0f;
  float mrow[8], lrow[8];
#pragma unroll
  for (int i = 0; i < 8; ++i) { mrow[i] = -1e30f; lrow[i] = 0.0f; }

  const int nS = t0 / 32 + 2;               // keys 0 .. t0+63 in 32-wide tiles
  for (int st = 0; st < nS; ++st) {
    const int s0 = st * 32;
    __syncthreads();
    // Stage K (transposed) and V: 32 keys x 256, 8 x b128 per thread each.
#pragma unroll
    for (int it = 0; it < 8; ++it) {
      int idx = it * 1024 + tid * 8;
      int key = idx >> 8, col = idx & 255;
      v8u kv = *(const v8u*)&kg[(size_t)(bz * T_ + s0 + key) * H_ + col];
#pragma unroll
      for (int e = 0; e < 8; ++e) KT[(col + e) * 32 + key] = kv[e];
      *(v8u*)&Vs[idx] = *(const v8u*)&vg[(size_t)(bz * T_ + s0 + key) * H_ + col];
    }
    __syncthreads();

    if (s0 <= wt0 + 15) {                   // wave-uniform: EXEC stays all-1s
      // logits: two 16x16 f32 tiles (s-local 0..15 and 16..31), contract H=256
      v8f c0, c1;
#pragma unroll
      for (int i = 0; i < 8; ++i) { c0[i] = 0.0f; c1[i] = 0.0f; }
#pragma unroll
      for (int hc = 0; hc < 8; ++hc) {
        BF16x16 af, b0, b1;
        af.h[0] = *(const v8bf*)&qrow[hc * 32 + 8 * hl];
        af.h[1] = *(const v8bf*)&qrow[hc * 32 + 16 + 8 * hl];
        const int kbase = hc * 32 * 32 + r;
#pragma unroll
        for (int s = 0; s < 16; ++s) {
          const int k = (s < 8) ? (8 * hl + s) : (16 + 8 * hl + (s - 8));
          b0.v[s] = *(const __bf16*)&KT[kbase + k * 32];
          b1.v[s] = *(const __bf16*)&KT[kbase + k * 32 + 16];
        }
        c0 = __builtin_amdgcn_wmma_f32_16x16x32_bf16(false, af.v, false, b0.v,
                                                     (short)0, c0, false, false);
        c1 = __builtin_amdgcn_wmma_f32_16x16x32_bf16(false, af.v, false, b1.v,
                                                     (short)0, c1, false, false);
      }

      // causal mask + online softmax (rows spread over 16-lane halves)
      float e0[8], e1[8], scl[8];
#pragma unroll
      for (int i = 0; i < 8; ++i) {
        const int tq = wt0 + 8 * hl + i;
        float v0 = (s0 + r      <= tq) ? c0[i] : -1e30f;
        float v1 = (s0 + 16 + r <= tq) ? c1[i] : -1e30f;
        float mx = fmaxf(v0, v1);
        mx = fmaxf(mx, __shfl_xor(mx, 1, 32));
        mx = fmaxf(mx, __shfl_xor(mx, 2, 32));
        mx = fmaxf(mx, __shfl_xor(mx, 4, 32));
        mx = fmaxf(mx, __shfl_xor(mx, 8, 32));
        float mnew = fmaxf(mrow[i], mx);
        float p0 = __expf(v0 - mnew);
        float p1 = __expf(v1 - mnew);
        float sum = p0 + p1;
        sum += __shfl_xor(sum, 1, 32);
        sum += __shfl_xor(sum, 2, 32);
        sum += __shfl_xor(sum, 4, 32);
        sum += __shfl_xor(sum, 8, 32);
        scl[i]  = __expf(mrow[i] - mnew);
        lrow[i] = lrow[i] * scl[i] + sum;
        mrow[i] = mnew;
        e0[i] = p0; e1[i] = p1;
      }
#pragma unroll
      for (int f = 0; f < 16; ++f)
#pragma unroll
        for (int i = 0; i < 8; ++i) O[f][i] *= scl[i];

      // P (C-layout) -> LDS bf16 -> reload as 16x32 A-fragment
#pragma unroll
      for (int i = 0; i < 8; ++i) {
        const int prow = i + 8 * hl;
        Pw[wave * 512 + prow * 32 + r]      = f2bf(e0[i]);
        Pw[wave * 512 + prow * 32 + 16 + r] = f2bf(e1[i]);
      }
      asm volatile("s_wait_dscnt 0" ::: "memory");
      BF16x16 pa;
      {
        const int pbase = wave * 512 + r * 32 + 8 * hl;
        pa.h[0] = *(const v8bf*)&Pw[pbase];
        pa.h[1] = *(const v8bf*)&Pw[pbase + 16];
      }
      // O += P (16x32) * V (32x256)
#pragma unroll
      for (int f = 0; f < 16; ++f) {
        BF16x16 vbf;
        const int vbase = f * 16 + r;
#pragma unroll
        for (int s = 0; s < 16; ++s) {
          const int k = (s < 8) ? (8 * hl + s) : (16 + 8 * hl + (s - 8));
          vbf.v[s] = *(const __bf16*)&Vs[vbase + k * 256];
        }
        O[f] = __builtin_amdgcn_wmma_f32_16x16x32_bf16(false, pa.v, false, vbf.v,
                                                       (short)0, O[f], false, false);
      }
    }
  }

  // normalize, store encoded (B*T, N*H) bf16
#pragma unroll
  for (int i = 0; i < 8; ++i) {
    const float inv = 1.0f / lrow[i];
    const size_t row = (size_t)(bz * T_ + wt0 + i + 8 * hl);
#pragma unroll
    for (int f = 0; f < 16; ++f)
      encg[row * NHD + n * H_ + f * 16 + r] = f2bf(O[f][i] * inv);
  }
}

// ---------------------------------------------------------------------------
// kernel_launch: orchestrate the 7-launch pipeline on `stream`.
// Workspace budget: ~115.4 MB (assumed <= ws_size).
// ---------------------------------------------------------------------------
extern "C" void kernel_launch(void* const* d_in, const int* in_sizes, int n_in,
                              void* d_out, int out_size, void* d_ws, size_t ws_size,
                              hipStream_t stream) {
  const float* x         = (const float*)d_in[0];
  const float* positions = (const float*)d_in[1];
  // d_in[2] = attn_mask (causal tril, reconstructed analytically in-kernel)
  const float* wq        = (const float*)d_in[3];
  const float* wkv       = (const float*)d_in[4];
  const float* wo        = (const float*)d_in[5];
  float* out             = (float*)d_out;

  char* ws = (char*)d_ws;
  unsigned short* xb   = (unsigned short*)(ws + 0);          // 16,777,216 B
  unsigned short* wcat = (unsigned short*)(ws + 16777216);   // 10,485,760 B
  unsigned short* wob  = (unsigned short*)(ws + 27262976);   //  8,388,608 B
  float*          qkvf = (float*)         (ws + 35651584);   // 41,943,040 B
  unsigned short* qb   = (unsigned short*)(ws + 77594624);   // 16,777,216 B
  unsigned short* kb   = (unsigned short*)(ws + 94371840);   //  2,097,152 B
  unsigned short* vb   = (unsigned short*)(ws + 96468992);   //  2,097,152 B
  unsigned short* encb = (unsigned short*)(ws + 98566144);   // 16,777,216 B
  (void)in_sizes; (void)n_in; (void)out_size; (void)ws_size;

  const int M = B_ * T_;  // 4096

  // 1) convert x and wo to bf16
  cvt_f32_bf16<<<(M * D_ / 4 + 255) / 256, 256, 0, stream>>>(x, xb, M * D_ / 4);
  cvt_f32_bf16<<<(NHD * D_ / 4 + 255) / 256, 256, 0, stream>>>(wo, wob, NHD * D_ / 4);

  // 2) build fused QKV weight (D x 2560)
  build_wcat<<<(D_ * QKV_COLS + 255) / 256, 256, 0, stream>>>(wq, wkv, wcat);

  // 3) fused QKV projection: (4096 x 2048) x (2048 x 2560) -> f32
  gemm_bf16_f32<<<dim3(QKV_COLS / 128, M / 64), 256, 0, stream>>>(
      xb, wcat, qkvf, M, QKV_COLS, D_);

  // 4) RoPE (+ H^-0.5 on q), split to bf16 q/k/v
  rope_split<<<M, 256, 0, stream>>>(qkvf, positions, qb, kb, vb);

  // 5) causal flash attention -> encoded bf16 (B*T, N*H)
  flash_attn<<<dim3(T_ / 64, NH_, B_), 128, 0, stream>>>(qb, kb, vb, encb);

  // 6) output projection + head-sum as one GEMM: (4096 x 2048) x (2048 x 2048)
  gemm_bf16_f32<<<dim3(D_ / 128, M / 64), 256, 0, stream>>>(
      encb, wob, out, M, D_, NHD);
}
